// HopfieldMemory_87196426043476
// MI455X (gfx1250) — compile-verified, matrix operand order checked by reading, not run on hardware
//
#include <hip/hip_runtime.h>
#include <math.h>

#define DIMN   1024
#define P0N    512
#define NHEAD  16
#define HDN    64
#define PTN    1536
#define BATCH  2
#define SEQ    1024
#define NEGI   (-1e9f)

typedef __attribute__((ext_vector_type(16))) __bf16        v16bf;
typedef __attribute__((ext_vector_type(8)))  float         v8f;
typedef __attribute__((ext_vector_type(4)))  unsigned int  u32x4;
typedef __attribute__((ext_vector_type(4)))  int           i32x4;

#if __has_builtin(__builtin_amdgcn_global_load_async_to_lds_b128) && \
    __has_builtin(__builtin_amdgcn_s_wait_asynccnt)
#define HAVE_ASYNC_LDS 1
typedef __attribute__((address_space(1))) i32x4 g_i32x4;   // global (__device__)
typedef __attribute__((address_space(3))) i32x4 l_i32x4;   // LDS (__shared__)
#endif

union FragB {
    v16bf          v;
    unsigned short u[16];
    u32x4          q[2];
};

__device__ __forceinline__ unsigned short f2bf(float f) {
    unsigned u = __builtin_bit_cast(unsigned, f);
    unsigned r = (u + 0x7FFFu + ((u >> 16) & 1u)) >> 16;   // round-to-nearest-even
    return (unsigned short)r;
}
__device__ __forceinline__ float b2f(unsigned short h) {
    return __builtin_bit_cast(float, ((unsigned)h) << 16);
}

__device__ __forceinline__ float wave_sum(float v) {
    #pragma unroll
    for (int o = 16; o; o >>= 1) v += __shfl_xor(v, o, 32);
    return v;
}
__device__ __forceinline__ float wave_max(float v) {
    #pragma unroll
    for (int o = 16; o; o >>= 1) v = fmaxf(v, __shfl_xor(v, o, 32));
    return v;
}

// Load a 16x32 bf16 row-major tile as a WMMA 16-bit A/B operand.
// Layout (ISA 7.12.2): lanes 0-15 (row m=lane)  : K 0..7 in q[0], K 16..23 in q[1]
//                      lanes 16-31 (row m=lane-16): K 8..15 in q[0], K 24..31 in q[1]
__device__ __forceinline__ void load_row_frag(FragB& f, const unsigned short* base,
                                              int ld, int lane) {
    const int half = lane >> 4;
    const int m    = lane & 15;
    const unsigned short* p = base + (size_t)m * ld + half * 8;
    f.q[0] = *(const u32x4*)(p);
    f.q[1] = *(const u32x4*)(p + 16);
}

__device__ __forceinline__ v8f wmma_bf16(const FragB& a, const FragB& b, v8f c) {
    return __builtin_amdgcn_wmma_f32_16x16x32_bf16(false, a.v, false, b.v,
                                                   (short)0, c, false, false);
}

// ---------------------------------------------------------------------------
// patterns = concat(stored, |z|) per batch, converted to bf16
// ---------------------------------------------------------------------------
__global__ void build_patterns_kernel(const float* __restrict__ zr,
                                      const float* __restrict__ zi,
                                      const float* __restrict__ stored,
                                      unsigned short* __restrict__ pat) {
    size_t i = (size_t)blockIdx.x * blockDim.x + threadIdx.x;
    const size_t total = (size_t)BATCH * PTN * DIMN;
    if (i >= total) return;
    int d = (int)(i % DIMN);
    int r = (int)((i / DIMN) % PTN);
    int b = (int)(i / ((size_t)PTN * DIMN));
    float val;
    if (r < P0N) {
        val = stored[(size_t)r * DIMN + d];
    } else {
        size_t o = ((size_t)b * SEQ + (r - P0N)) * DIMN + d;
        float a = zr[o], c = zi[o];
        val = sqrtf(a * a + c * c);
    }
    pat[i] = f2bf(val);
}

__global__ void cvt_kernel(const float* __restrict__ src,
                           unsigned short* __restrict__ dst, int n) {
    int i = blockIdx.x * blockDim.x + threadIdx.x;
    if (i < n) dst[i] = f2bf(src[i]);
}

// ---------------------------------------------------------------------------
// Core: one wave computes a 32x32 tile of C = A(M,K) @ W(N,K)^T
// 2 A-frags x 2 B-frags -> 4 accumulators: 8 b128 loads per 4 WMMA
// ---------------------------------------------------------------------------
__device__ __forceinline__ void gemm_core_32x32(const unsigned short* arow0,
                                                const unsigned short* brow0,
                                                int K, int lane, v8f acc[4]) {
    const unsigned short* arow1 = arow0 + (size_t)16 * K;
    const unsigned short* brow1 = brow0 + (size_t)16 * K;
    for (int kk = 0; kk < K; kk += 32) {
        FragB a0, a1, b0, b1;
        load_row_frag(a0, arow0 + kk, K, lane);
        load_row_frag(a1, arow1 + kk, K, lane);
        load_row_frag(b0, brow0 + kk, K, lane);
        load_row_frag(b1, brow1 + kk, K, lane);
        acc[0] = wmma_bf16(a0, b0, acc[0]);
        acc[1] = wmma_bf16(a0, b1, acc[1]);
        acc[2] = wmma_bf16(a1, b0, acc[2]);
        acc[3] = wmma_bf16(a1, b1, acc[3]);
    }
}

// bf16 output variant (q/k/v projections, V^T)
__global__ void __launch_bounds__(128)
gemm_bf16_kernel(const unsigned short* __restrict__ A,
                 const unsigned short* __restrict__ W,
                 unsigned short* __restrict__ C,
                 int M, int N, int K, long aStride, long wStride, long cStride) {
    const int lane = threadIdx.x & 31;
    const int wave = threadIdx.x >> 5;
    A += (size_t)blockIdx.y * aStride;
    W += (size_t)blockIdx.y * wStride;
    C += (size_t)blockIdx.y * cStride;
    const int tilesN = N >> 5;
    const int tiles  = (M >> 5) * tilesN;
    const int t = blockIdx.x * 4 + wave;
    if (t >= tiles) return;                    // wave-uniform: EXEC stays full
    const int tm = t / tilesN, tn = t % tilesN;
    v8f acc[4] = {v8f{}, v8f{}, v8f{}, v8f{}};
    gemm_core_32x32(A + (size_t)(tm * 32) * K, W + (size_t)(tn * 32) * K, K, lane, acc);
    const int half = lane >> 4, n = lane & 15;
    #pragma unroll
    for (int blk = 0; blk < 4; ++blk) {
        const int bm = blk >> 1, bn = blk & 1;
        unsigned short* cp = C + (size_t)(tm * 32 + bm * 16) * N + tn * 32 + bn * 16 + n;
        #pragma unroll
        for (int j = 0; j < 8; ++j) {
            int m = j + half * 8;
            cp[(size_t)m * N] = f2bf(acc[blk][j]);
        }
    }
}

// fp32-out + bias variant (output projection)
__global__ void __launch_bounds__(128)
gemm_f32_bias_kernel(const unsigned short* __restrict__ A,
                     const unsigned short* __restrict__ W,
                     const float* __restrict__ bias,
                     float* __restrict__ C,
                     int M, int N, int K, long aStride, long cStride) {
    const int lane = threadIdx.x & 31;
    const int wave = threadIdx.x >> 5;
    A += (size_t)blockIdx.y * aStride;
    C += (size_t)blockIdx.y * cStride;
    const int tilesN = N >> 5;
    const int tiles  = (M >> 5) * tilesN;
    const int t = blockIdx.x * 4 + wave;
    if (t >= tiles) return;
    const int tm = t / tilesN, tn = t % tilesN;
    v8f acc[4] = {v8f{}, v8f{}, v8f{}, v8f{}};
    gemm_core_32x32(A + (size_t)(tm * 32) * K, W + (size_t)(tn * 32) * K, K, lane, acc);
    const int half = lane >> 4, n = lane & 15;
    #pragma unroll
    for (int blk = 0; blk < 4; ++blk) {
        const int bm = blk >> 1, bn = blk & 1;
        const float bv = bias[tn * 32 + bn * 16 + n];
        float* cp = C + (size_t)(tm * 32 + bm * 16) * N + tn * 32 + bn * 16 + n;
        #pragma unroll
        for (int j = 0; j < 8; ++j) {
            int m = j + half * 8;
            cp[(size_t)m * N] = acc[blk][j] + bv;
        }
    }
}

// ---------------------------------------------------------------------------
// One Hopfield iteration for a (batch, head, 16-row s-tile):
//   state tile  -> LDS via global_load_async_to_lds_b128 (ASYNCcnt path)
//   scores = state.k^T * inv_scale + mask   (WMMA, bf16 rows into LDS)
//   sparsemax via tau bisection             (wave shuffle reductions)
//   state  = weights @ V (via V^T rows)     (WMMA)
// block = 4 waves; LDS = 16x1536 bf16 scores (48KB) + 16x64 state tile + taus
// ---------------------------------------------------------------------------
__global__ void __launch_bounds__(128)
attn_step_kernel(const unsigned short* __restrict__ state_in,
                 const unsigned short* __restrict__ kmat,
                 const unsigned short* __restrict__ vtmat,   // V^T: (B, DIMN, PTN)
                 const float* __restrict__ log_temp,
                 unsigned short* __restrict__ state_out) {
    __shared__ __align__(16) unsigned short sc[16 * PTN];
    __shared__ __align__(16) unsigned short stile[16 * HDN];
    __shared__ float tauS[16];

    const int lane = threadIdx.x & 31;
    const int wave = threadIdx.x >> 5;
    const int s0   = blockIdx.x * 16;
    const int h    = blockIdx.y;
    const int b    = blockIdx.z;
    const int half = lane >> 4;
    const int nl   = lane & 15;

    const float lt        = fminf(fmaxf(log_temp[h], -4.f), 4.f);
    const float inv_scale = 1.0f / (8.0f * expf(lt));   // 1/(sqrt(64)*temp)

    const unsigned short* sbase  = state_in + ((size_t)b * SEQ + s0) * DIMN + h * HDN;
    const unsigned short* kbase  = kmat + (size_t)b * PTN * DIMN + h * HDN;
    const unsigned short* vtbase = vtmat + (size_t)b * DIMN * PTN + (size_t)(h * HDN) * PTN;

    // ---- stage state tile (16 x 64 bf16 = 2KB) into LDS, once per block ----
    {
        const int t   = (int)threadIdx.x;        // 0..127, 16B chunk each
        const int e   = t * 8;                   // element offset in tile
        const int row = e >> 6, col = e & 63;
        const unsigned short* gsrc = sbase + (size_t)row * DIMN + col;
        unsigned short*       ldst = &stile[e];
#ifdef HAVE_ASYNC_LDS
        __builtin_amdgcn_global_load_async_to_lds_b128(
            (g_i32x4*)gsrc, (l_i32x4*)ldst, 0, 0);
        __builtin_amdgcn_s_wait_asynccnt(0);
#else
        *(u32x4*)ldst = *(const u32x4*)gsrc;
#endif
    }
    __syncthreads();

    // ---- phase 1: scores ----
    FragB a0, a1;
    load_row_frag(a0, stile,      HDN, lane);
    load_row_frag(a1, stile + 32, HDN, lane);

    for (int pt = wave; pt < PTN / 16; pt += 4) {
        const unsigned short* kb = kbase + (size_t)(pt * 16) * DIMN;
        if (pt + 4 < PTN / 16)
            __builtin_prefetch(kbase + (size_t)((pt + 4) * 16) * DIMN, 0, 1);
        FragB b0, b1;
        load_row_frag(b0, kb,      DIMN, lane);
        load_row_frag(b1, kb + 32, DIMN, lane);
        v8f acc = {};
        acc = wmma_bf16(a0, b0, acc);
        acc = wmma_bf16(a1, b1, acc);
        const int p = pt * 16 + nl;
        #pragma unroll
        for (int j = 0; j < 8; ++j) {
            int m = j + half * 8;
            float mval = (p < P0N) ? 0.f : (((p - P0N) <= (s0 + m)) ? 0.f : NEGI);
            sc[m * PTN + p] = f2bf(acc[j] * inv_scale + mval);
        }
    }
    __syncthreads();

    // ---- phase 2: sparsemax threshold tau per row (bisection) ----
    for (int r = wave; r < 16; r += 4) {
        const unsigned short* row = &sc[r * PTN];
        float zmax = -3.4e38f;
        for (int p = lane; p < PTN; p += 32) zmax = fmaxf(zmax, b2f(row[p]));
        zmax = wave_max(zmax);
        float lo = zmax - 1.0f, hi = zmax;
        for (int it = 0; it < 30; ++it) {
            float mid = 0.5f * (lo + hi);
            float s = 0.f;
            for (int p = lane; p < PTN; p += 32) s += fmaxf(b2f(row[p]) - mid, 0.f);
            s = wave_sum(s);
            if (s > 1.0f) lo = mid; else hi = mid;
        }
        if (lane == 0) tauS[r] = 0.5f * (lo + hi);
    }
    __syncthreads();

    // ---- phase 3: state = weights @ V; V^T rows are contiguous b128 loads ----
    // wave w produces head cols [16w, 16w+16)
    const float tau_m = tauS[lane & 15];     // A-operand row for this lane
    const unsigned short* vtw = vtbase + (size_t)(wave * 16) * PTN;
    v8f acc = {};
    for (int p0 = 0; p0 < PTN; p0 += 32) {
        FragB a, bb;
        {   // weights fragment from LDS scores (clip(z - tau, 0)), on-the-fly bf16
            const unsigned short* wp = &sc[(lane & 15) * PTN + p0 + half * 8];
            FragB raw;
            raw.q[0] = *(const u32x4*)(wp);
            raw.q[1] = *(const u32x4*)(wp + 16);
            #pragma unroll
            for (int j = 0; j < 16; ++j)
                a.u[j] = f2bf(fmaxf(b2f(raw.u[j]) - tau_m, 0.f));
        }
        load_row_frag(bb, vtw + p0, PTN, lane);   // rows = head cols, K = p
        acc = wmma_bf16(a, bb, acc);
    }
    unsigned short* outp = state_out + ((size_t)b * SEQ + s0) * DIMN + h * HDN + wave * 16 + nl;
    #pragma unroll
    for (int j = 0; j < 8; ++j) {
        int m = j + half * 8;
        outp[(size_t)m * DIMN] = f2bf(acc[j]);
    }
}

// ---------------------------------------------------------------------------
// LayerNorm over last dim (2*DIM), one block per row
// ---------------------------------------------------------------------------
__global__ void __launch_bounds__(256)
layernorm_kernel(const float* __restrict__ x, const float* __restrict__ gamma,
                 const float* __restrict__ beta, float* __restrict__ out) {
    __shared__ float r1[256], r2[256];
    const int row = blockIdx.x;
    const int C = 2 * DIMN;
    const float* xr = x + (size_t)row * C;
    float s1 = 0.f, s2 = 0.f;
    for (int c = threadIdx.x; c < C; c += 256) { float v = xr[c]; s1 += v; s2 += v * v; }
    r1[threadIdx.x] = s1; r2[threadIdx.x] = s2;
    __syncthreads();
    for (int o = 128; o; o >>= 1) {
        if (threadIdx.x < o) { r1[threadIdx.x] += r1[threadIdx.x + o];
                               r2[threadIdx.x] += r2[threadIdx.x + o]; }
        __syncthreads();
    }
    const float mu   = r1[0] / C;
    const float var  = r2[0] / C - mu * mu;
    const float rstd = rsqrtf(var + 1e-5f);
    float* orow = out + (size_t)row * C;
    for (int c = threadIdx.x; c < C; c += 256)
        orow[c] = (xr[c] - mu) * rstd * gamma[c] + beta[c];
}

// ---------------------------------------------------------------------------
extern "C" void kernel_launch(void* const* d_in, const int* in_sizes, int n_in,
                              void* d_out, int out_size, void* d_ws, size_t ws_size,
                              hipStream_t stream) {
    const float* zr       = (const float*)d_in[0];
    const float* zi       = (const float*)d_in[1];
    /* d_in[2] = mask: reconstructed analytically */
    const float* stored   = (const float*)d_in[3];
    const float* Wq       = (const float*)d_in[4];
    const float* Wk       = (const float*)d_in[5];
    const float* Wv       = (const float*)d_in[6];
    const float* Wo       = (const float*)d_in[7];
    const float* bo       = (const float*)d_in[8];
    const float* log_temp = (const float*)d_in[9];
    const float* gamma    = (const float*)d_in[10];
    const float* beta     = (const float*)d_in[11];
    float* out = (float*)d_out;

    char* ws = (char*)d_ws;
    size_t off = 0;
    auto alloc = [&](size_t bytes) -> char* {
        char* p = ws + off;
        off += (bytes + 255) & ~(size_t)255;
        return p;
    };
    unsigned short* pat    = (unsigned short*)alloc((size_t)BATCH * PTN * DIMN * 2);
    unsigned short* Wqb    = (unsigned short*)alloc((size_t)DIMN * DIMN * 2);
    unsigned short* Wkb    = (unsigned short*)alloc((size_t)DIMN * DIMN * 2);
    unsigned short* Wvb    = (unsigned short*)alloc((size_t)DIMN * DIMN * 2);
    unsigned short* Wob    = (unsigned short*)alloc((size_t)2 * DIMN * DIMN * 2);
    unsigned short* kbuf   = (unsigned short*)alloc((size_t)BATCH * PTN * DIMN * 2);
    unsigned short* vtbuf  = (unsigned short*)alloc((size_t)BATCH * DIMN * PTN * 2);
    unsigned short* stateA = (unsigned short*)alloc((size_t)BATCH * SEQ * DIMN * 2);
    unsigned short* stateB = (unsigned short*)alloc((size_t)BATCH * SEQ * DIMN * 2);
    float*          outpre = (float*)alloc((size_t)BATCH * SEQ * 2 * DIMN * 4);

    // patterns (stored || |z|) -> bf16
    {
        size_t total = (size_t)BATCH * PTN * DIMN;
        build_patterns_kernel<<<(unsigned)((total + 255) / 256), 256, 0, stream>>>(zr, zi, stored, pat);
    }
    // weight conversions
    cvt_kernel<<<(DIMN * DIMN + 255) / 256, 256, 0, stream>>>(Wq, Wqb, DIMN * DIMN);
    cvt_kernel<<<(DIMN * DIMN + 255) / 256, 256, 0, stream>>>(Wk, Wkb, DIMN * DIMN);
    cvt_kernel<<<(DIMN * DIMN + 255) / 256, 256, 0, stream>>>(Wv, Wvb, DIMN * DIMN);
    cvt_kernel<<<(2 * DIMN * DIMN + 255) / 256, 256, 0, stream>>>(Wo, Wob, 2 * DIMN * DIMN);

    // q = mag @ Wq^T  -> initial state (A = dynamic rows of patterns)
    {
        int tiles = (SEQ / 32) * (DIMN / 32);
        gemm_bf16_kernel<<<dim3((tiles + 3) / 4, BATCH), 128, 0, stream>>>(
            pat + (size_t)P0N * DIMN, Wqb, stateA, SEQ, DIMN, DIMN,
            (long)PTN * DIMN, 0L, (long)SEQ * DIMN);
    }
    // k = patterns @ Wk^T   (row-major K)
    {
        int tiles = (PTN / 32) * (DIMN / 32);
        gemm_bf16_kernel<<<dim3((tiles + 3) / 4, BATCH), 128, 0, stream>>>(
            pat, Wkb, kbuf, PTN, DIMN, DIMN, (long)PTN * DIMN, 0L, (long)PTN * DIMN);
    }
    // V^T directly: VT(D, PT) = Wv(D,K) x patterns(PT,K)^T  (swap operand roles)
    {
        int tiles = (DIMN / 32) * (PTN / 32);
        gemm_bf16_kernel<<<dim3((tiles + 3) / 4, BATCH), 128, 0, stream>>>(
            Wvb, pat, vtbuf, DIMN, PTN, DIMN, 0L, (long)PTN * DIMN, (long)DIMN * PTN);
    }
    // 3 Hopfield iterations (ping-pong state)
    unsigned short* sin  = stateA;
    unsigned short* sout = stateB;
    for (int step = 0; step < 3; ++step) {
        attn_step_kernel<<<dim3(SEQ / 16, NHEAD, BATCH), 128, 0, stream>>>(
            sin, kbuf, vtbuf, log_temp, sout);
        unsigned short* t = sin; sin = sout; sout = t;
    }
    // output projection + bias (final state is in `sin` after last swap)
    {
        int tiles = (SEQ / 32) * (2 * DIMN / 32);
        gemm_f32_bias_kernel<<<dim3((tiles + 3) / 4, BATCH), 128, 0, stream>>>(
            sin, Wob, bo, outpre, SEQ, 2 * DIMN, DIMN,
            (long)SEQ * DIMN, (long)SEQ * 2 * DIMN);
    }
    // LayerNorm -> d_out
    layernorm_kernel<<<BATCH * SEQ, 256, 0, stream>>>(outpre, gamma, beta, out);
}
